// MolGraphPrior_19628000542885
// MI455X (gfx1250) — matrix operand order repaired and checked by reading, not compiled, not run on hardware
//
#include <hip/hip_runtime.h>
#include <hip/hip_bf16.h>

typedef __bf16 bf16_t;
typedef __attribute__((ext_vector_type(16))) __bf16 v16bf;
typedef __attribute__((ext_vector_type(8)))  float  v8f;

#define HID 256
#define EPSV 1e-5f

// ---------------------------------------------------------------- utilities
__global__ void k_zero_f32(float* p, long n) {
  long i = (long)blockIdx.x * blockDim.x + threadIdx.x;
  if (i < n) p[i] = 0.f;
}

// out = relu(X[N,Kin] @ W[Kin,256] + b); writes f32 and/or bf16 (null to skip)
__global__ void k_proj_relu(const float* __restrict__ X, const float* __restrict__ W,
                            const float* __restrict__ b, float* __restrict__ outF,
                            bf16_t* __restrict__ outB, int Kin) {
  __shared__ float xr[512];
  int n = blockIdx.x, c = threadIdx.x;
  for (int k = c; k < Kin; k += blockDim.x) xr[k] = X[(size_t)n * Kin + k];
  __syncthreads();
  float acc = b[c];
  for (int k = 0; k < Kin; ++k) acc += xr[k] * W[(size_t)k * HID + c];
  acc = fmaxf(acc, 0.f);
  if (outF) outF[(size_t)n * HID + c] = acc;
  if (outB) outB[(size_t)n * HID + c] = (bf16_t)acc;
}

// one wave (32 lanes) per edge; lane handles 8 channels.
// agg[dst] += relu(x[src] + e[edge])   (f32 atomics into L2)
__global__ void k_msg_scatter(const float* __restrict__ x, const bf16_t* __restrict__ e,
                              const int* __restrict__ src, const int* __restrict__ dst,
                              float* __restrict__ agg, int E) {
  int gw = (int)((blockIdx.x * blockDim.x + threadIdx.x) >> 5);
  int lane = threadIdx.x & 31;
  if (gw >= E) return;
  int s = src[gw], d = dst[gw];
  const float* xp = x + (size_t)s * HID + lane * 8;
  const bf16_t* ep = e + (size_t)gw * HID + lane * 8;
  float* ap = agg + (size_t)d * HID + lane * 8;
  float4 x0 = *(const float4*)xp;
  float4 x1 = *(const float4*)(xp + 4);
  uint4 eq = *(const uint4*)ep;
  float m[8] = {x0.x, x0.y, x0.z, x0.w, x1.x, x1.y, x1.z, x1.w};
  unsigned ew[4] = {eq.x, eq.y, eq.z, eq.w};
#pragma unroll
  for (int j = 0; j < 4; ++j) {
    float e0 = __uint_as_float(ew[j] << 16);
    float e1 = __uint_as_float(ew[j] & 0xffff0000u);
    float v0 = fmaxf(m[2 * j] + e0, 0.f);
    float v1 = fmaxf(m[2 * j + 1] + e1, 0.f);
    atomicAdd(ap + 2 * j, v0);
    atomicAdd(ap + 2 * j + 1, v1);
  }
}

// h_bf16 = bf16(x + agg)
__global__ void k_hprep(const float* __restrict__ x, const float* __restrict__ agg,
                        bf16_t* __restrict__ h, long n) {
  long i = (long)blockIdx.x * blockDim.x + threadIdx.x;
  if (i < n) h[i] = (bf16_t)(x[i] + agg[i]);
}

// Pack f32 weight W[K,N] into WMMA B-fragment layout:
// P[((nt*(K/32)+kc)*32 + lane)*16 + i] = W[kc*32 + (lane<16?0:16) + i][nt*16 + (lane&15)]
__global__ void k_pack_b(const float* __restrict__ W, bf16_t* __restrict__ P, int K, int N) {
  long t = (long)blockIdx.x * blockDim.x + threadIdx.x;
  long total = (long)K * N;
  if (t >= total) return;
  int i = (int)(t & 15);
  int lane = (int)((t >> 4) & 31);
  long rest = t >> 9;
  int kchunks = K >> 5;
  int kc = (int)(rest % kchunks);
  int nt = (int)(rest / kchunks);
  int k = kc * 32 + ((lane < 16) ? 0 : 16) + i;
  int n = nt * 16 + (lane & 15);
  P[t] = (bf16_t)W[(size_t)k * N + n];
}

// C[M,N] = A[M,K](bf16) @ Bpack(bf16) + bias, optional relu, out bf16 or f32.
// block = 256 threads = 8 waves; each wave owns TWO adjacent 16x16 column tiles
// (block covers 256 output columns). The 16xK A panel is staged in LDS once
// (single barrier), with +16B row padding so half-wave fragment reads spread
// across all 64 LDS banks. K-loop is barrier-free: 2 ds_load_b128 (A frag) +
// 4 global_load_b128 (B frags) + 2 WMMAs per 32-chunk.
template <int K, bool RELU, bool OUTB>
__global__ void k_gemm_wmma(const bf16_t* __restrict__ A, const bf16_t* __restrict__ Bp,
                            const float* __restrict__ bias, float* __restrict__ outF,
                            bf16_t* __restrict__ outB, int N) {
  constexpr int STR = K + 8;          // padded LDS row stride (elements)
  constexpr int NC  = K / 32;         // k-chunks
  __shared__ __align__(16) bf16_t apanel[16 * STR];
  const int tid  = threadIdx.x;
  const int lane = tid & 31;
  const int wv   = tid >> 5;
  const int rowBase = blockIdx.x * 16;

  // cooperative A panel load: 16 rows x K, 8 bf16 (16B) per thread per iter
#pragma unroll
  for (int it = 0; it < (16 * K) / (256 * 8); ++it) {
    int e   = (it * 256 + tid) * 8;
    int row = e / K;
    int kp  = e % K;
    *(uint4*)&apanel[row * STR + kp] =
        *(const uint4*)&A[(size_t)(rowBase + row) * K + kp];
  }
  __syncthreads();

  const int r  = lane & 15;
  const int ko = (lane < 16) ? 0 : 16;
  const bf16_t* abase = &apanel[r * STR + ko];
  const int nt0 = blockIdx.y * 16 + wv * 2;

  v8f acc0 = {}, acc1 = {};
#pragma unroll 4
  for (int kc = 0; kc < NC; ++kc) {
    union { v16bf v; uint4 q[2]; } afr, bf0, bf1;
    const bf16_t* ap = abase + kc * 32;
    afr.q[0] = *(const uint4*)ap;
    afr.q[1] = *(const uint4*)(ap + 8);
    const bf16_t* p0 = Bp + (((size_t)nt0 * NC + kc) * 32 + lane) * 16;
    const bf16_t* p1 = Bp + (((size_t)(nt0 + 1) * NC + kc) * 32 + lane) * 16;
    bf0.q[0] = *(const uint4*)p0;
    bf0.q[1] = *(const uint4*)(p0 + 8);
    bf1.q[0] = *(const uint4*)p1;
    bf1.q[1] = *(const uint4*)(p1 + 8);
    acc0 = __builtin_amdgcn_wmma_f32_16x16x32_bf16(false, afr.v, false, bf0.v,
                                                   (short)0, acc0, false, false);
    acc1 = __builtin_amdgcn_wmma_f32_16x16x32_bf16(false, afr.v, false, bf1.v,
                                                   (short)0, acc1, false, false);
  }

  const int col0 = nt0 * 16 + (lane & 15);
  const float bv0 = bias[col0];
  const float bv1 = bias[col0 + 16];
#pragma unroll
  for (int rr = 0; rr < 8; ++rr) {
    int row = rowBase + rr + ((lane < 16) ? 0 : 8);
    float v0 = acc0[rr] + bv0;
    float v1 = acc1[rr] + bv1;
    if (RELU) { v0 = fmaxf(v0, 0.f); v1 = fmaxf(v1, 0.f); }
    if (OUTB) {
      outB[(size_t)row * N + col0]      = (bf16_t)v0;
      outB[(size_t)row * N + col0 + 16] = (bf16_t)v1;
    } else {
      outF[(size_t)row * N + col0]      = v0;
      outF[(size_t)row * N + col0 + 16] = v1;
    }
  }
}

// per-channel sum / sumsq partials, one atomic pair per (block, channel)
__global__ void k_bn_stats(const float* __restrict__ h2, float* __restrict__ stats, int N) {
  int c = threadIdx.x;
  int r0 = blockIdx.x * 256;
  int r1 = min(r0 + 256, N);
  float s = 0.f, ss = 0.f;
  for (int r = r0; r < r1; ++r) {
    float v = h2[(size_t)r * HID + c];
    s += v; ss += v * v;
  }
  atomicAdd(&stats[c], s);
  atomicAdd(&stats[HID + c], ss);
}

__global__ void k_bn_apply(const float* __restrict__ h2, const float* __restrict__ stats,
                           const float* __restrict__ gamma, const float* __restrict__ beta,
                           float* __restrict__ x, int N) {
  long i = (long)blockIdx.x * blockDim.x + threadIdx.x;
  long n = (long)N * HID;
  if (i >= n) return;
  int c = (int)(i & (HID - 1));
  float invN = 1.f / (float)N;
  float mu = stats[c] * invN;
  float var = stats[HID + c] * invN - mu * mu;
  float v = gamma[c] * (h2[i] - mu) * rsqrtf(var + EPSV) + beta[c];
  x[i] = fmaxf(v, 0.f);
}

__global__ void k_pool_acc(const float* __restrict__ x, const int* __restrict__ batch,
                           float* __restrict__ pool, float* __restrict__ cnt) {
  int n = blockIdx.x, c = threadIdx.x;
  int g = batch[n];
  atomicAdd(&pool[(size_t)g * HID + c], x[(size_t)n * HID + c]);
  if (c == 0) atomicAdd(&cnt[g], 1.f);
}

__global__ void k_pool_div(float* __restrict__ pool, const float* __restrict__ cnt) {
  int g = blockIdx.x, c = threadIdx.x;
  pool[(size_t)g * HID + c] /= fmaxf(cnt[g], 1.f);
}

__global__ void k_go_scatter(const float* __restrict__ goh, const int* __restrict__ tgt,
                             const int* __restrict__ gidx, float* __restrict__ agg,
                             float* __restrict__ cnt, int E) {
  int e = blockIdx.x, c = threadIdx.x;
  if (e >= E) return;
  int t = tgt[e], g = gidx[e];
  atomicAdd(&agg[(size_t)t * HID + c], goh[(size_t)g * HID + c]);
  if (c == 0) atomicAdd(&cnt[t], 1.f);
}

__global__ void k_build_f(const float* __restrict__ pemb, const float* __restrict__ goagg,
                          const float* __restrict__ cnt, float* __restrict__ f,
                          float* __restrict__ present) {
  int p = blockIdx.x, c = threadIdx.x;
  f[(size_t)p * 512 + c] = pemb[(size_t)p * HID + c];
  f[(size_t)p * 512 + HID + c] = goagg[(size_t)p * HID + c] / fmaxf(cnt[p], 1.f);
  if (c == 0) present[p] = (cnt[p] > 0.f) ? 1.f : 0.f;
}

// out[m,c] = op(A[m,:K] @ W[K,256] + b)  (row in LDS; small M only)
__global__ void k_row_mlp(const float* __restrict__ A, const float* __restrict__ W,
                          const float* __restrict__ b, float* __restrict__ out,
                          int K, int relu) {
  __shared__ float ar[512];
  int m = blockIdx.x, c = threadIdx.x;
  for (int k = c; k < K; k += blockDim.x) ar[k] = A[(size_t)m * K + k];
  __syncthreads();
  float acc = b[c];
  for (int k = 0; k < K; ++k) acc += ar[k] * W[(size_t)k * HID + c];
  if (relu) acc = fmaxf(acc, 0.f);
  out[(size_t)m * HID + c] = acc;
}

// h = pemb + present*fused; per-row LayerNorm -> out
__global__ void k_go_final(const float* __restrict__ pemb, const float* __restrict__ fused,
                           const float* __restrict__ present, const float* __restrict__ lng,
                           const float* __restrict__ lnb, float* __restrict__ out) {
  __shared__ float red[256];
  int p = blockIdx.x, c = threadIdx.x;
  float h = pemb[(size_t)p * HID + c] + present[p] * fused[(size_t)p * HID + c];
  red[c] = h; __syncthreads();
  for (int s = 128; s > 0; s >>= 1) { if (c < s) red[c] += red[c + s]; __syncthreads(); }
  float mu = red[0] / (float)HID;
  __syncthreads();
  float d = h - mu;
  red[c] = d * d; __syncthreads();
  for (int s = 128; s > 0; s >>= 1) { if (c < s) red[c] += red[c + s]; __syncthreads(); }
  float var = red[0] / (float)HID;
  out[(size_t)p * HID + c] = lng[c] * d * rsqrtf(var + EPSV) + lnb[c];
}

// ---------------------------------------------------------------- host side
struct EncP {
  const float *We, *Wn, *be, *bn;
  const float *W1[4], *W2[4], *b1[4], *b2[4], *beta[4], *gamma[4];
};

static void load_enc(void* const* d_in, int base, EncP& p) {
  p.We = (const float*)d_in[base + 0];
  p.Wn = (const float*)d_in[base + 1];
  p.be = (const float*)d_in[base + 2];
  p.bn = (const float*)d_in[base + 3];
  for (int l = 0; l < 4; ++l) {
    int i = base + 4 + 6 * l;
    p.W1[l]    = (const float*)d_in[i + 0];
    p.W2[l]    = (const float*)d_in[i + 1];
    p.b1[l]    = (const float*)d_in[i + 2];
    p.b2[l]    = (const float*)d_in[i + 3];
    p.beta[l]  = (const float*)d_in[i + 4];
    p.gamma[l] = (const float*)d_in[i + 5];
  }
}

extern "C" void kernel_launch(void* const* d_in, const int* in_sizes, int n_in,
                              void* d_out, int out_size, void* d_ws, size_t ws_size,
                              hipStream_t stream) {
  (void)in_sizes; (void)n_in; (void)out_size; (void)ws_size;
  const int NP_N = 102400, NP_E = 409600, N_PROT = 512;
  const int ND_N = 65536,  ND_E = 131072, N_DRUG = 4096;
  const int N_GO = 5000,   GO_DIM = 200,  PG_E   = 15360;

  const float* prot_x  = (const float*)d_in[0];
  const int*   prot_ei = (const int*)  d_in[1];
  const float* prot_ea = (const float*)d_in[2];
  const int*   prot_b  = (const int*)  d_in[3];
  const float* drug_x  = (const float*)d_in[4];
  const int*   drug_ei = (const int*)  d_in[5];
  const float* drug_ea = (const float*)d_in[6];
  const int*   drug_b  = (const int*)  d_in[7];
  const int*   pg_ei   = (const int*)  d_in[8];

  EncP PE, DE;
  load_enc(d_in, 11, PE);
  load_enc(d_in, 39, DE);
  const float* goW1 = (const float*)d_in[67];
  const float* goW2 = (const float*)d_in[68];
  const float* goWg = (const float*)d_in[69];
  const float* gob1 = (const float*)d_in[70];
  const float* gob2 = (const float*)d_in[71];
  const float* gobg = (const float*)d_in[72];
  const float* goE  = (const float*)d_in[73];
  const float* lnb  = (const float*)d_in[74];
  const float* lng  = (const float*)d_in[75];

  // bump allocator over workspace (buffers sized for the larger protein graph,
  // reused for the drug graph which runs second)
  char* ws = (char*)d_ws;
  size_t off = 0;
  auto alloc = [&](size_t bytes) -> void* {
    void* p = ws + off;
    off += (bytes + 255) & ~(size_t)255;
    return p;
  };
  float*  xbuf  = (float*) alloc((size_t)NP_N * HID * 4);
  bf16_t* ebuf  = (bf16_t*)alloc((size_t)NP_E * HID * 2);
  float*  aggb  = (float*) alloc((size_t)NP_N * HID * 4);
  bf16_t* hbuf  = (bf16_t*)alloc((size_t)NP_N * HID * 2);
  bf16_t* z1buf = (bf16_t*)alloc((size_t)NP_N * 512 * 2);
  float*  h2buf = (float*) alloc((size_t)NP_N * HID * 4);
  float*  stats = (float*) alloc(512 * 4);
  bf16_t* wp1P[4], *wp2P[4], *wp1D[4], *wp2D[4];
  for (int l = 0; l < 4; ++l) { wp1P[l] = (bf16_t*)alloc(256 * 512 * 2); wp2P[l] = (bf16_t*)alloc(512 * 256 * 2); }
  for (int l = 0; l < 4; ++l) { wp1D[l] = (bf16_t*)alloc(256 * 512 * 2); wp2D[l] = (bf16_t*)alloc(512 * 256 * 2); }
  float* p_raw   = (float*)alloc((size_t)N_PROT * HID * 4);
  float* cntP    = (float*)alloc(N_PROT * 4);
  float* cntD    = (float*)alloc(N_DRUG * 4);
  float* gohbuf  = (float*)alloc((size_t)N_GO * HID * 4);
  float* goagg   = (float*)alloc((size_t)N_PROT * HID * 4);
  float* gocnt   = (float*)alloc(N_PROT * 4);
  float* present = (float*)alloc(N_PROT * 4);
  float* fbuf    = (float*)alloc((size_t)N_PROT * 512 * 4);
  float* fused1  = (float*)alloc((size_t)N_PROT * HID * 4);
  float* fused2  = (float*)alloc((size_t)N_PROT * HID * 4);

  float* dout = (float*)d_out;
  float* poolD = dout + (size_t)N_PROT * HID;  // drug means written straight to output

  // pack all MLP weights to bf16 WMMA fragment layout (once per call)
  for (int l = 0; l < 4; ++l) {
    k_pack_b<<<512, 256, 0, stream>>>(PE.W1[l], wp1P[l], 256, 512);
    k_pack_b<<<512, 256, 0, stream>>>(PE.W2[l], wp2P[l], 512, 256);
    k_pack_b<<<512, 256, 0, stream>>>(DE.W1[l], wp1D[l], 256, 512);
    k_pack_b<<<512, 256, 0, stream>>>(DE.W2[l], wp2D[l], 512, 256);
  }

  auto run_enc = [&](const float* X, int Kn, const int* ei, const float* EA, int Ke,
                     const int* batch, int N, int E, int G, const EncP& P,
                     bf16_t* const* wp1, bf16_t* const* wp2, float* pool, float* cnt) {
    long nElem = (long)N * HID;
    k_proj_relu<<<N, 256, 0, stream>>>(X, P.Wn, P.bn, xbuf, nullptr, Kn);
    k_proj_relu<<<E, 256, 0, stream>>>(EA, P.We, P.be, nullptr, ebuf, Ke);
    for (int l = 0; l < 4; ++l) {
      k_zero_f32<<<(unsigned)((nElem + 255) / 256), 256, 0, stream>>>(aggb, nElem);
      k_msg_scatter<<<E / 8, 256, 0, stream>>>(xbuf, ebuf, ei, ei + E, aggb, E);
      k_hprep<<<(unsigned)((nElem + 255) / 256), 256, 0, stream>>>(xbuf, aggb, hbuf, nElem);
      // z1 = relu(h @ W1 + b1)  [N,512] bf16   (K=256, 2 col-blocks of 256)
      k_gemm_wmma<256, true, true><<<dim3(N / 16, 2), 256, 0, stream>>>(
          hbuf, wp1[l], P.b1[l], nullptr, z1buf, 512);
      // h2 = z1 @ W2 + b2       [N,256] f32    (K=512, 1 col-block of 256)
      k_gemm_wmma<512, false, false><<<dim3(N / 16, 1), 256, 0, stream>>>(
          z1buf, wp2[l], P.b2[l], h2buf, nullptr, 256);
      k_zero_f32<<<2, 256, 0, stream>>>(stats, 512);
      k_bn_stats<<<(N + 255) / 256, 256, 0, stream>>>(h2buf, stats, N);
      k_bn_apply<<<(unsigned)((nElem + 255) / 256), 256, 0, stream>>>(h2buf, stats, P.gamma[l], P.beta[l], xbuf, N);
    }
    long pElem = (long)G * HID;
    k_zero_f32<<<(unsigned)((pElem + 255) / 256), 256, 0, stream>>>(pool, pElem);
    k_zero_f32<<<(G + 255) / 256, 256, 0, stream>>>(cnt, G);
    k_pool_acc<<<N, 256, 0, stream>>>(xbuf, batch, pool, cnt);
    k_pool_div<<<G, 256, 0, stream>>>(pool, cnt);
  };

  // protein encoder -> p_raw (ws), then drug encoder -> d_out[131072:]
  run_enc(prot_x, 11, prot_ei, prot_ea, 13, prot_b, NP_N, NP_E, N_PROT, PE, wp1P, wp2P, p_raw, cntP);
  run_enc(drug_x, 20, drug_ei, drug_ea, 5, drug_b, ND_N, ND_E, N_DRUG, DE, wp1D, wp2D, poolD, cntD);

  // GO enrichment -> d_out[0:131072]
  k_proj_relu<<<N_GO, 256, 0, stream>>>(goE, goWg, gobg, gohbuf, nullptr, GO_DIM);
  long gElem = (long)N_PROT * HID;
  k_zero_f32<<<(unsigned)((gElem + 255) / 256), 256, 0, stream>>>(goagg, gElem);
  k_zero_f32<<<(N_PROT + 255) / 256, 256, 0, stream>>>(gocnt, N_PROT);
  k_go_scatter<<<PG_E, 256, 0, stream>>>(gohbuf, pg_ei, pg_ei + PG_E, goagg, gocnt, PG_E);
  k_build_f<<<N_PROT, 256, 0, stream>>>(p_raw, goagg, gocnt, fbuf, present);
  k_row_mlp<<<N_PROT, 256, 0, stream>>>(fbuf, goW1, gob1, fused1, 512, 1);
  k_row_mlp<<<N_PROT, 256, 0, stream>>>(fused1, goW2, gob2, fused2, 256, 0);
  k_go_final<<<N_PROT, 256, 0, stream>>>(p_raw, fused2, present, lng, lnb, dout);
}
